// LSTMLayer_85194971283618
// MI455X (gfx1250) — compile-verified, matrix operand order checked by reading, not run on hardware
//
#include <hip/hip_runtime.h>

// ---------------------------------------------------------------------------
// LSTM layer, T=2048, B=64, I=H=512 (gate dim G=2048) for MI455X / gfx1250.
// Persistent-grid scan, gate-aligned tiling:
//   block b: batch rows rt=b>>4 (16 rows), h-columns jb=b&15 (32 cols).
//   Its 8 waves compute the 8 gate tiles {i,f,g,o} x {2 col tiles} for that
//   patch, so the LSTM pointwise update is block-local:
//     - gate exchange via 8 KB LDS (one __syncthreads, no global gates buf)
//     - cell state c lives in registers for the whole 2048-step scan
//     - ONE device-wide barrier per step (h broadcast only)
// bf16 WMMA 16x16x32 with fp32 accumulation for both GEMMs:
//   - W_hh fragments register-resident (128 VGPRs/lane)
//   - W_ih slices staged once to LDS via async global->LDS DMA
//   - x_t tile double-buffered in LDS, prefetched one step ahead
//     (counted s_wait_asynccnt 2 keeps the prefetch in flight)
//   - h tile staged per step via async DMA after the grid barrier
// x_t staged as bf16 inside its own d_out slot (consumed before overwrite).
// ---------------------------------------------------------------------------

#define TT   2048
#define BB   64
#define II   512
#define HH   512
#define GG   2048            // 4*H
#define BH   (BB * HH)       // 32768
#define NBLK 64              // persistent grid blocks (8 waves each)

typedef unsigned short u16;
typedef unsigned int   u32;

typedef __attribute__((ext_vector_type(16))) __bf16 v16bf;
typedef __attribute__((ext_vector_type(8)))  __bf16 v8bf;
typedef __attribute__((ext_vector_type(8)))  float  v8f;

// ---- LDS layout. Padded rows: 512 bf16 + 8 pad = 1040 B (bank-conflict-free:
// start bank of a b128 = 4n+4hi mod 64, each half-wave covers all 64 banks). --
#define ROWB     1040
#define WIH_LDS  (8 * 16 * ROWB)              // 133120: per-wave W_ih slices
#define XS0_OFF  WIH_LDS                      // x ping
#define XS1_OFF  (WIH_LDS + 16 * ROWB)        // x pong
#define HS_OFF   (WIH_LDS + 32 * ROWB)        // h tile
#define GL_OFF   (WIH_LDS + 48 * ROWB)        // gate exchange [4][16][32] f32
#define SMEM_BYTES (GL_OFF + 4 * 16 * 32 * 4) // 191232 < 320 KB

// ---- ws layout (byte offsets) ---------------------------------------------
#define OFF_SYNC 0u
#define OFF_H    0x10000u    // h state, bf16 [64][512]   (64 KB)
#define OFF_WIH  0x100000u   // W_ih bf16 [2048][512]     (2 MB)
#define OFF_WHH  0x300000u   // W_hh bf16 [2048][512]     (2 MB)

extern __shared__ __align__(16) char smem[];

__device__ __forceinline__ u16 f2bf(float f) {
  union { float f; u32 u; } v; v.f = f;
  u32 r = v.u + 0x7FFFu + ((v.u >> 16) & 1u);   // round-to-nearest-even
  return (u16)(r >> 16);
}

__device__ __forceinline__ float sigm(float x) { return 1.0f / (1.0f + __expf(-x)); }
__device__ __forceinline__ float tanh_f(float x) {
  float e = __expf(-2.0f * x);
  return (1.0f - e) / (1.0f + e);
}

__device__ __forceinline__ v16bf mk16(v8bf a, v8bf b) {
  v16bf r;
#pragma unroll
  for (int i = 0; i < 8; ++i) { r[i] = a[i]; r[i + 8] = b[i]; }
  return r;
}

// Async global -> LDS copy of 16 bytes per lane (ASYNCcnt-tracked, GV mode).
__device__ __forceinline__ void async_g2l16(u32 lds_off, const void* gaddr) {
  asm volatile("global_load_async_to_lds_b128 %0, %1, off"
               :: "v"(lds_off), "v"((unsigned long long)(uintptr_t)gaddr)
               : "memory");
}
__device__ __forceinline__ void async_wait0() {
  asm volatile("s_wait_asynccnt 0" ::: "memory");
}
__device__ __forceinline__ void async_wait2() {   // allow 2 in flight (x prefetch)
  asm volatile("s_wait_asynccnt 2" ::: "memory");
}

// ---------------------------------------------------------------------------
// Prepass: zero barrier counter; bf16-convert weights / h0 / inp.
// inp[t] (bf16) is staged in the first 64 KB of d_out's 128 KB slot for t.
// ---------------------------------------------------------------------------
__global__ void lstm_prepass(const float* __restrict__ inp,
                             const float* __restrict__ h0,
                             const float* __restrict__ wih,
                             const float* __restrict__ whh,
                             float* __restrict__ out,
                             u16*  __restrict__ wihb,
                             u16*  __restrict__ whhb,
                             u16*  __restrict__ hbuf,
                             u32*  __restrict__ sync) {
  size_t tid = (size_t)blockIdx.x * blockDim.x + threadIdx.x;
  size_t nth = (size_t)gridDim.x * blockDim.x;
  if (tid == 0) *sync = 0u;

  const size_t NW = (size_t)GG * II;            // 1,048,576 per weight matrix
  for (size_t i = tid; i < NW; i += nth) {
    wihb[i] = f2bf(wih[i]);
    whhb[i] = f2bf(whh[i]);
  }
  for (size_t i = tid; i < (size_t)BH; i += nth) hbuf[i] = f2bf(h0[i]);

  const size_t NX = (size_t)TT * BH;            // 67,108,864
  for (size_t i = tid; i < NX; i += nth) {
    size_t t = i >> 15;                         // / 32768 (B*I)
    size_t e = i & 32767;
    u16* slot = (u16*)((char*)out + t * (size_t)(BH * 4));
    slot[e] = f2bf(inp[i]);
  }
}

// ---------------------------------------------------------------------------
// Persistent scan kernel.
// ---------------------------------------------------------------------------
__global__ __launch_bounds__(256)
void lstm_scan(float* __restrict__ out,
               const u16* __restrict__ wihb,
               const u16* __restrict__ whhb,
               const float* __restrict__ b_ih,
               const float* __restrict__ b_hh,
               const float* __restrict__ c0,
               u16*  __restrict__ hbuf,
               u32*  __restrict__ sync) {
  const int lane = threadIdx.x & 31;
  const int wv   = threadIdx.x >> 5;            // wave in block 0..7
  const int rt   = blockIdx.x >> 4;             // batch row tile 0..3
  const int jb   = blockIdx.x & 15;             // h-column group (32 cols)
  const int gt   = wv >> 1;                     // gate 0..3 (i,f,g,o)
  const int jl   = wv & 1;                      // column sub-tile 0..1
  const int n    = lane & 15;
  const int hi   = lane >> 4;
  // This wave's gate-matrix column (also its W row).
  const int gcol  = gt * 512 + (jb * 2 + jl) * 16 + n;
  const int grow0 = gcol - n;                   // first of 16 weight rows

  const float biasv = b_ih[gcol] + b_hh[gcol];

  // Preload W_hh B-fragments: 16 x v16bf = 128 VGPRs, resident for all steps.
  const u16* wrow_hh = whhb + (size_t)gcol * II;
  v16bf Bhh[16];
#pragma unroll
  for (int kk = 0; kk < 16; ++kk) {
    const int klo = kk * 32 + hi * 8;
    Bhh[kk] = mk16(*(const v8bf*)(wrow_hh + klo),
                   *(const v8bf*)(wrow_hh + klo + 16));
  }

  // Stage cols 0..255 of this wave's 16 W_ih rows into LDS once (async DMA, 8 KB/wave).
  const u32 wbase = (u32)wv * (16 * ROWB);
  {
    const u16* gsl = wihb + (size_t)grow0 * II;
#pragma unroll
    for (int k = 0; k < 16; ++k) {
      const int id  = lane + 32 * k;            // 0..511 chunks of 16 B
      const int row = id >> 5, cc = id & 31;
      async_g2l16(wbase + (u32)(row * ROWB + cc * 16),
                  gsl + (size_t)row * II + cc * 8);
    }
  }
  // Stage x_0 into ping buffer (completion enforced by the loop-top wait).
  {
    const u16* xsl = (const u16*)out + (size_t)(rt * 16) * II;
#pragma unroll
    for (int k = 0; k < 2; ++k) {
      const int id  = threadIdx.x + 256 * k;
      const int row = id >> 5, cc = id & 31;
      async_g2l16((u32)(XS0_OFF + row * ROWB + cc * 16),
                  xsl + (size_t)row * II + cc * 8);
    }
  }

  const char* wrow_l = smem + wbase + n * ROWB;
  const char* hrow_l = smem + HS_OFF + n * ROWB;

  // Per-thread LSTM state: 2 of the block's 512 (row, col) elements.
  float creg[2];
  int   gidx[2], erow[2], ecol[2];
#pragma unroll
  for (int k = 0; k < 2; ++k) {
    const int e = threadIdx.x + 256 * k;        // 0..511
    erow[k] = e >> 5;                           // 0..15
    ecol[k] = e & 31;                           // 0..31
    gidx[k] = (rt * 16 + erow[k]) * HH + jb * 32 + ecol[k];
    creg[k] = c0[gidx[k]];
  }

  u32 epoch = 0;

  for (int t = 0; t < TT; ++t) {
    // ---- stage h_{t-1} (always) and prefetch x_{t+1} into the idle buffer --
    {
      const u16* hsl = hbuf + (size_t)(rt * 16) * HH;
#pragma unroll
      for (int k = 0; k < 2; ++k) {
        const int id  = threadIdx.x + 256 * k;
        const int row = id >> 5, cc = id & 31;
        async_g2l16((u32)(HS_OFF + row * ROWB + cc * 16),
                    hsl + (size_t)row * HH + cc * 8);
      }
    }
    if (t + 1 < TT) {
      const u32 xoff = ((t + 1) & 1) ? XS1_OFF : XS0_OFF;
      const u16* xsl = (const u16*)((const char*)out + (size_t)(t + 1) * (BH * 4))
                       + (size_t)(rt * 16) * II;
#pragma unroll
      for (int k = 0; k < 2; ++k) {
        const int id  = threadIdx.x + 256 * k;
        const int row = id >> 5, cc = id & 31;
        async_g2l16(xoff + (u32)(row * ROWB + cc * 16),
                    xsl + (size_t)row * II + cc * 8);
      }
      async_wait2();   // in-order: h tile + current x done; prefetch in flight
    } else {
      async_wait0();
    }
    __syncthreads();

    const char* xrow_l = smem + ((t & 1) ? XS1_OFF : XS0_OFF) + n * ROWB;

    v8f C;
#pragma unroll
    for (int r = 0; r < 8; ++r) C[r] = biasv;

    // x_t @ W_ih^T  (A and B from LDS, conflict-free ds_load_b128)
#pragma unroll
    for (int kk = 0; kk < 16; ++kk) {
      const int off = kk * 64 + hi * 16;
      v16bf A  = mk16(*(const v8bf*)(xrow_l + off),
                      *(const v8bf*)(xrow_l + off + 32));
      v16bf Bw = mk16(*(const v8bf*)(wrow_l + off),
                      *(const v8bf*)(wrow_l + off + 32));
      C = __builtin_amdgcn_wmma_f32_16x16x32_bf16(false, A, false, Bw,
                                                  (short)0, C, false, false);
    }
    // h_{t-1} @ W_hh^T  (A from LDS, B register-resident)
#pragma unroll
    for (int kk = 0; kk < 16; ++kk) {
      const int off = kk * 64 + hi * 16;
      v16bf A = mk16(*(const v8bf*)(hrow_l + off),
                     *(const v8bf*)(hrow_l + off + 32));
      C = __builtin_amdgcn_wmma_f32_16x16x32_bf16(false, A, false, Bhh[kk],
                                                  (short)0, C, false, false);
    }

    // ---- gate exchange in LDS: gl[gate][row 0..15][col 0..31] f32 ----
    // C layout: VGPR r, lane l -> M = r + 8*(l>>4), N = l&15.
#pragma unroll
    for (int r = 0; r < 8; ++r) {
      const int row = hi * 8 + r;
      *(float*)(smem + GL_OFF + gt * 2048 + row * 128 + (jl * 16 + n) * 4) = C[r];
    }
    __syncthreads();

    // ---- block-local LSTM pointwise update (c in registers) ----
#pragma unroll
    for (int k = 0; k < 2; ++k) {
      const char* gbase = smem + GL_OFF + erow[k] * 128 + ecol[k] * 4;
      const float ig = sigm(*(const float*)(gbase));
      const float fg = sigm(*(const float*)(gbase + 2048));
      const float gg = tanh_f(*(const float*)(gbase + 4096));
      const float og = sigm(*(const float*)(gbase + 6144));
      const float cn = fg * creg[k] + ig * gg;
      const float hn = og * tanh_f(cn);
      creg[k] = cn;
      out[(size_t)t * BH + gidx[k]] = hn;   // overwrites x_t staging (consumed)
      hbuf[gidx[k]] = f2bf(hn);
      if (t == TT - 1) {
        out[(size_t)TT * BH + gidx[k]]      = hn;   // h_T
        out[(size_t)TT * BH + BH + gidx[k]] = cn;   // c_T
      }
    }

    // ---- single grid barrier per step: h_t visible device-wide ----
    __threadfence();
    __syncthreads();
    if (threadIdx.x == 0) {
      __hip_atomic_fetch_add(sync, 1u, __ATOMIC_RELEASE, __HIP_MEMORY_SCOPE_AGENT);
      while (__hip_atomic_load(sync, __ATOMIC_ACQUIRE, __HIP_MEMORY_SCOPE_AGENT)
             < epoch + NBLK)
        __builtin_amdgcn_s_sleep(1);
    }
    __syncthreads();
    epoch += NBLK;
  }
}

// ---------------------------------------------------------------------------
extern "C" void kernel_launch(void* const* d_in, const int* in_sizes, int n_in,
                              void* d_out, int out_size, void* d_ws, size_t ws_size,
                              hipStream_t stream) {
  const float* inp  = (const float*)d_in[0];
  const float* h0   = (const float*)d_in[1];
  const float* c0   = (const float*)d_in[2];
  const float* W_ih = (const float*)d_in[3];
  const float* W_hh = (const float*)d_in[4];
  const float* b_ih = (const float*)d_in[5];
  const float* b_hh = (const float*)d_in[6];
  float* out = (float*)d_out;

  char* ws = (char*)d_ws;
  u32* sync = (u32*)(ws + OFF_SYNC);
  u16* hbuf = (u16*)(ws + OFF_H);
  u16* wihb = (u16*)(ws + OFF_WIH);
  u16* whhb = (u16*)(ws + OFF_WHH);

  (void)in_sizes; (void)n_in; (void)out_size; (void)ws_size;

  // Allow 191,232 B of dynamic LDS (not a stream op; capture-safe).
  (void)hipFuncSetAttribute((const void*)lstm_scan,
                            hipFuncAttributeMaxDynamicSharedMemorySize,
                            SMEM_BYTES);

  lstm_prepass<<<4096, 256, 0, stream>>>(inp, h0, W_ih, W_hh, out,
                                         wihb, whhb, hbuf, sync);
  lstm_scan<<<NBLK, 256, SMEM_BYTES, stream>>>(out, wihb, whhb, b_ih, b_hh,
                                               c0, hbuf, sync);
}
